// GroupedQueryAttention_38062000177661
// MI455X (gfx1250) — matrix operand order repaired and checked by reading, not compiled, not run on hardware
//
#include <hip/hip_runtime.h>

typedef __attribute__((ext_vector_type(16))) __bf16 v16bf;
typedef __attribute__((ext_vector_type(8)))  float  v8f;
typedef __attribute__((ext_vector_type(8)))  __bf16 v8bf;
typedef __attribute__((ext_vector_type(4)))  float  vf4;
typedef __attribute__((ext_vector_type(4)))  __bf16 vbf4;

constexpr int Bn = 2, Hn = 8, Gn = 4, Ln = 2048, Sn = 2048, Dn = 128;
constexpr int TS = 64;                // S tile width
constexpr int NT = Sn / TS;           // 32 S tiles
constexpr int WAVES = 8;
constexpr int ROWS = 16;              // L rows per wave
constexpr int BLOCK_L = WAVES * ROWS; // 128

union Frag { v16bf v; uint4 u[2]; };
union Pk8  { v8bf v; uint4 u; };
union Pk4  { vbf4 v; uint2 u; };

// Load one K tile (row-major bf16 [s][d]) and one V tile (transposed bf16 [d][s]).
__device__ __forceinline__ void load_kv_tile(unsigned short* kt, unsigned short* vt,
                                             const float* kBase, const float* vBase,
                                             int st, int tid) {
  // K: thread -> row = tid>>2 (0..63), 32 d-columns
  {
    const int row = tid >> 2;
    const int cb  = (tid & 3) * 32;
    const vf4* kr = (const vf4*)(kBase + (size_t)(st * TS + row) * Dn + cb);
    #pragma unroll
    for (int j = 0; j < 8; ++j) {
      Pk4 p; p.v = __builtin_convertvector(kr[j], vbf4);
      *(uint2*)&kt[row * Dn + cb + j * 4] = p.u;
    }
  }
  // V transpose: thread -> 4x8 block (4 s-rows x 8 d-cols)
  {
    const int sb = (tid >> 4) * 4;   // 0..60
    const int db = (tid & 15) * 8;   // 0..120
    vf4 f[4][2];
    #pragma unroll
    for (int i = 0; i < 4; ++i) {
      const vf4* vr = (const vf4*)(vBase + (size_t)(st * TS + sb + i) * Dn + db);
      f[i][0] = vr[0];
      f[i][1] = vr[1];
    }
    #pragma unroll
    for (int h = 0; h < 2; ++h)
      #pragma unroll
      for (int e = 0; e < 4; ++e) {
        vf4 col = { f[0][h][e], f[1][h][e], f[2][h][e], f[3][h][e] };
        Pk4 p; p.v = __builtin_convertvector(col, vbf4);
        *(uint2*)&vt[(db + h * 4 + e) * TS + sb] = p.u;
      }
  }
}

__global__ __launch_bounds__(256) void gqa_fa_kernel(
    const float* __restrict__ q, const float* __restrict__ k,
    const float* __restrict__ v, float* __restrict__ out) {
  __shared__ unsigned short Kt[2][TS * Dn];        // [s][d]  16KB each
  __shared__ unsigned short Vt[2][Dn * TS];        // [d][s]  16KB each
  __shared__ unsigned short Pst[WAVES][ROWS * TS]; // per-wave P^T / Q staging

  const int tid  = threadIdx.x;
  const int wave = tid >> 5;
  const int lane = tid & 31;
  const int lh   = lane & 15;
  const int hi   = lane >> 4;

  const int LT  = Ln / BLOCK_L;           // 16
  const int bid = blockIdx.x;
  const int lt  = bid % LT;
  const int bhg = bid / LT;               // ((b*H+h)*G+g)
  const int bh  = bhg / Gn;               // K/V broadcast over G

  const float* qBase = q + ((size_t)bhg * Ln + (size_t)lt * BLOCK_L + wave * ROWS) * Dn;
  const float* kBase = k + (size_t)bh * Sn * Dn;
  const float* vBase = v + (size_t)bh * Sn * Dn;
  float*       oBase = out + ((size_t)bhg * Ln + (size_t)lt * BLOCK_L + wave * ROWS) * Dn;

  unsigned short* pst = &Pst[wave][0];

  // ---- Stage Q fp32->bf16 per wave; build 4 B fragments of Q^T (d-slices of 32) ----
  // B layout: lanes 0-15 = column q=lane with K=d 0..7,16..23; lanes 16-31 = q=lane-16, K=d 8..15,24..31.
  v16bf qb[4];
  #pragma unroll
  for (int p = 0; p < 4; ++p) {
    const vf4* qr = (const vf4*)(qBase + (size_t)lh * Dn + p * 32 + hi * 16);
    #pragma unroll
    for (int j = 0; j < 4; ++j) {
      Pk4 pk; pk.v = __builtin_convertvector(qr[j], vbf4);
      *(uint2*)&pst[lh * 32 + hi * 16 + j * 4] = pk.u;   // flat 16x32 staging
    }
    Frag b;                                   // in-order DS within the wave
    const int off = lh * 32 + hi * 8;
    b.u[0] = *(const uint4*)&pst[off];
    b.u[1] = *(const uint4*)&pst[off + 16];
    qb[p] = b.v;
  }

  // ---- Online-softmax state: one (m,l) per lane = per query row ----
  const v8f zf = {};
  float m_run = -1e30f, l_run = 0.f;
  v8f of[8];                                  // O^T accumulators: M=d tile dj, N=lane=q
  #pragma unroll
  for (int dj = 0; dj < 8; ++dj) of[dj] = zf;

  load_kv_tile(Kt[0], Vt[0], kBase, vBase, 0, tid);
  __syncthreads();

  for (int st = 0; st < NT; ++st) {
    const int cur = st & 1;
    if (st + 1 < NT)
      load_kv_tile(Kt[cur ^ 1], Vt[cur ^ 1], kBase, vBase, st + 1, tid);

    const unsigned short* kt = Kt[cur];
    const unsigned short* vt = Vt[cur];

    // ---- S^T = K * Q^T : A = K tile rows (s), B = Q^T (resident) ----
    v8f sc[4];
    #pragma unroll
    for (int m = 0; m < 4; ++m) sc[m] = zf;
    #pragma unroll
    for (int m = 0; m < 4; ++m) {
      #pragma unroll
      for (int kk = 0; kk < 4; ++kk) {
        Frag a;                                            // A: row=s=m*16+lh, d chunks
        const int off = (m * 16 + lh) * Dn + kk * 32 + hi * 8;
        a.u[0] = *(const uint4*)&kt[off];
        a.u[1] = *(const uint4*)&kt[off + 16];
        sc[m] = __builtin_amdgcn_wmma_f32_16x16x32_bf16(
            false, a.v, false, qb[kk], (short)0, sc[m], false, false);
      }
    }

    // ---- online softmax, column-wise == per lane; only 2 cross-lane shuffles ----
    float mx = fmaxf(fmaxf(sc[0][0], sc[1][0]), fmaxf(sc[2][0], sc[3][0]));
    #pragma unroll
    for (int e = 1; e < 8; ++e)
      mx = fmaxf(mx, fmaxf(fmaxf(sc[0][e], sc[1][e]), fmaxf(sc[2][e], sc[3][e])));
    mx = fmaxf(mx, __shfl_xor(mx, 16, 32));   // combine the two s-halves
    const float mn   = fmaxf(m_run, mx);
    const float corr = __expf(m_run - mn);
    m_run = mn;
    float ls = 0.f;
    #pragma unroll
    for (int m = 0; m < 4; ++m)
      #pragma unroll
      for (int e = 0; e < 8; ++e) {
        float pe = __expf(sc[m][e] - mn);
        sc[m][e] = pe;
        ls += pe;
      }
    ls += __shfl_xor(ls, 16, 32);
    l_run = l_run * corr + ls;
    #pragma unroll
    for (int dj = 0; dj < 8; ++dj)
      #pragma unroll
      for (int e = 0; e < 8; ++e) of[dj][e] *= corr;

    // ---- P^T -> per-wave LDS: 8 contiguous bf16 per lane per fragment ----
    #pragma unroll
    for (int m = 0; m < 4; ++m) {
      Pk8 p; p.v = __builtin_convertvector(sc[m], v8bf);
      *(uint4*)&pst[lh * TS + m * 16 + hi * 8] = p.u;   // Ptt[q=lh][s]
    }

    // ---- O^T += V^T * P^T : A = V^T rows (d), B = P^T (two s-slices) ----
    #pragma unroll
    for (int ss = 0; ss < 2; ++ss) {
      Frag bp;
      const int offb = lh * TS + ss * 32 + hi * 8;  // B: column q=lh, K=s chunk
      bp.u[0] = *(const uint4*)&pst[offb];
      bp.u[1] = *(const uint4*)&pst[offb + 16];
      #pragma unroll
      for (int dj = 0; dj < 8; ++dj) {
        Frag a;                                // A: row d=dj*16+lh, s chunks
        const int offa = (dj * 16 + lh) * TS + ss * 32 + hi * 8;
        a.u[0] = *(const uint4*)&vt[offa];
        a.u[1] = *(const uint4*)&vt[offa + 16];
        of[dj] = __builtin_amdgcn_wmma_f32_16x16x32_bf16(
            false, a.v, false, bp.v, (short)0, of[dj], false, false);
      }
    }

    __syncthreads();
  }

  // ---- epilogue: O^T/l, contiguous d per lane -> float4 stores ----
  const float inv = 1.0f / l_run;
  #pragma unroll
  for (int dj = 0; dj < 8; ++dj) {
    vf4 o0 = { of[dj][0] * inv, of[dj][1] * inv, of[dj][2] * inv, of[dj][3] * inv };
    vf4 o1 = { of[dj][4] * inv, of[dj][5] * inv, of[dj][6] * inv, of[dj][7] * inv };
    float* dst = oBase + (size_t)lh * Dn + dj * 16 + hi * 8;
    *(vf4*)(dst)     = o0;
    *(vf4*)(dst + 4) = o1;
  }
}

extern "C" void kernel_launch(void* const* d_in, const int* in_sizes, int n_in,
                              void* d_out, int out_size, void* d_ws, size_t ws_size,
                              hipStream_t stream) {
  (void)in_sizes; (void)n_in; (void)out_size; (void)d_ws; (void)ws_size;
  const float* q = (const float*)d_in[0];
  const float* k = (const float*)d_in[1];
  const float* v = (const float*)d_in[2];
  const int grid = Bn * Hn * Gn * (Ln / BLOCK_L);  // 1024 blocks
  gqa_fa_kernel<<<grid, 256, 0, stream>>>(q, k, v, (float*)d_out);
}
